// VectorQuantizer_42339787604802
// MI455X (gfx1250) — compile-verified, matrix operand order checked by reading, not run on hardware
//
#include <hip/hip_runtime.h>

typedef __attribute__((ext_vector_type(16))) _Float16 v16h;
typedef __attribute__((ext_vector_type(8)))  float    v8f;

#define NBATCH 32
#define DIM 64
#define HW 4096                     // 64*64 pixels per (b,c) plane
#define NPIX (NBATCH * HW)          // 131072 pixel vectors
#define KCODES 512
#define PIX_PER_WG 128
#define NWG (NPIX / PIX_PER_WG)     // 1024
#define OUT_ELEMS (NPIX * DIM)      // 8388608

// LDS padding to spread banks (strides keep 16B alignment for wide ds loads)
#define EMB_STRIDE 72               // halves per codebook row
#define X_STRIDE   68               // floats per pixel row

__global__ __launch_bounds__(256)
void vq_fused(const float* __restrict__ inp,   // [32,64,64,64] NCHW
              const float* __restrict__ emb,   // [512,64] f32
              float* __restrict__ out,         // [32,64,64,64] NCHW quantized
              float* __restrict__ idxOut,      // [131072] indices as float
              float* __restrict__ partial)     // [1024] loss partials
{
    __shared__ _Float16 sEmb[KCODES * EMB_STRIDE];   // 73,728 B
    __shared__ float    sX[PIX_PER_WG * X_STRIDE];   // 34,816 B
    __shared__ float    sE2[KCODES];                 //  2,048 B (reused as loss scratch)
    __shared__ int      sBest[PIX_PER_WG];           //    512 B

    const int tid  = threadIdx.x;
    const int lane = tid & 31;
    const int wave = tid >> 5;
    const int blk  = blockIdx.x;

    const int g0  = blk * PIX_PER_WG;   // first flat pixel of this tile
    const int b   = g0 >> 12;           // 4096 pixels per image
    const int hw0 = g0 & (HW - 1);      // multiple of 128 -> tile stays in one image

    // ---- Stage x tile: transpose NCHW plane slices into [pixel][c] f32 (coalesced)
    #pragma unroll
    for (int i = 0; i < 32; ++i) {
        int linear = tid + 256 * i;           // 0..8191
        int c = linear >> 7;
        int p = linear & (PIX_PER_WG - 1);
        float v = inp[((b * DIM + c) << 12) + hw0 + p];
        sX[p * X_STRIDE + c] = v;
    }

    // ---- Stage codebook as f16 + per-code squared norms (f32)
    #pragma unroll
    for (int rep = 0; rep < 2; ++rep) {
        int code = tid + 256 * rep;
        const float* er = emb + code * DIM;
        float ss = 0.f;
        #pragma unroll
        for (int d = 0; d < DIM; ++d) {
            float v = er[d];
            ss = fmaf(v, v, ss);
            sEmb[code * EMB_STRIDE + d] = (_Float16)v;
        }
        sE2[code] = ss;
    }
    __syncthreads();

    // ---- Build A fragments for this wave's 16 rows (held for all 32 code tiles)
    const int m       = lane & 15;
    const int halfSel = lane >> 4;
    const int row     = wave * 16 + m;
    const float* xr   = &sX[row * X_STRIDE];

    v16h a0, a1;
    #pragma unroll
    for (int j = 0; j < 8; ++j) {
        a0[j]     = (_Float16)xr[halfSel * 8 + j];
        a0[8 + j] = (_Float16)xr[16 + halfSel * 8 + j];
        a1[j]     = (_Float16)xr[32 + halfSel * 8 + j];
        a1[8 + j] = (_Float16)xr[48 + halfSel * 8 + j];
    }

    float bestv[8];
    int   besti[8];
    #pragma unroll
    for (int r = 0; r < 8; ++r) { bestv[r] = 3.4e38f; besti[r] = 0; }

    // ---- Distance panel: 16 rows x 512 codes via 32 tiles x 2 WMMAs.
    // Unroll x2 so two independent tiles' ds-loads / WMMA chains / argmin
    // tails software-pipeline (VALU tail co-executes with XDL WMMAs).
    #pragma unroll 2
    for (int t0 = 0; t0 < 32; ++t0) {
        const int n0  = t0 * 16;
        const int col = n0 + m;
        const _Float16* ebr = &sEmb[col * EMB_STRIDE];

        v16h b0, b1;
        #pragma unroll
        for (int j = 0; j < 8; ++j) {
            b0[j]     = ebr[halfSel * 8 + j];
            b0[8 + j] = ebr[16 + halfSel * 8 + j];
            b1[j]     = ebr[32 + halfSel * 8 + j];
            b1[8 + j] = ebr[48 + halfSel * 8 + j];
        }

        v8f acc = {};
        acc = __builtin_amdgcn_wmma_f32_16x16x32_f16(false, a0, false, b0,
                                                     (short)0, acc, false, false);
        acc = __builtin_amdgcn_wmma_f32_16x16x32_f16(false, a1, false, b1,
                                                     (short)0, acc, false, false);

        const float e2v = sE2[col];
        #pragma unroll
        for (int r = 0; r < 8; ++r) {
            float d = fmaf(-2.f, acc[r], e2v);  // ||e||^2 - 2 x.e  (||x||^2 const/row)
            if (d < bestv[r]) { bestv[r] = d; besti[r] = col; }
        }
    }

    // ---- Cross-lane argmin within each 16-lane group (first-min tie-break)
    #pragma unroll
    for (int r = 0; r < 8; ++r) {
        float bv = bestv[r];
        int   bi = besti[r];
        #pragma unroll
        for (int mask = 8; mask >= 1; mask >>= 1) {
            float ov = __shfl_xor(bv, mask, 32);
            int   oi = __shfl_xor(bi, mask, 32);
            if (ov < bv || (ov == bv && oi < bi)) { bv = ov; bi = oi; }
        }
        if (m == 0) sBest[wave * 16 + halfSel * 8 + r] = bi;
    }
    __syncthreads();

    // ---- Indices output (as float, reference returns int argmin values)
    if (tid < PIX_PER_WG) idxOut[g0 + tid] = (float)sBest[tid];

    // ---- Gather + NCHW store + loss partial (straight-through out == quantized)
    const int p     = tid & (PIX_PER_WG - 1);
    const int chalf = tid >> 7;
    const int hw    = hw0 + p;
    const float* qrow = emb + sBest[p] * DIM;

    float lsum = 0.f;
    #pragma unroll
    for (int cc = 0; cc < 32; ++cc) {
        int c = cc * 2 + chalf;
        float q  = qrow[c];
        float xv = sX[p * X_STRIDE + c];
        out[((b * DIM + c) << 12) + hw] = q;
        float dd = q - xv;
        lsum = fmaf(dd, dd, lsum);
    }

    // ---- Deterministic block reduction of loss partial (reuse sE2 as scratch)
    sE2[tid] = lsum;
    __syncthreads();
    #pragma unroll
    for (int s = 128; s >= 1; s >>= 1) {
        if (tid < s) sE2[tid] += sE2[tid + s];
        __syncthreads();
    }
    if (tid == 0) partial[blk] = sE2[0];
}

__global__ __launch_bounds__(256)
void vq_loss(const float* __restrict__ partial, float* __restrict__ lossOut)
{
    __shared__ float red[256];
    int tid = threadIdx.x;
    float s = 0.f;
    for (int i = tid; i < NWG; i += 256) s += partial[i];
    red[tid] = s;
    __syncthreads();
    #pragma unroll
    for (int st = 128; st >= 1; st >>= 1) {
        if (tid < st) red[tid] += red[tid + st];
        __syncthreads();
    }
    if (tid == 0) lossOut[0] = 0.25f * red[0] / (float)OUT_ELEMS;
}

extern "C" void kernel_launch(void* const* d_in, const int* in_sizes, int n_in,
                              void* d_out, int out_size, void* d_ws, size_t ws_size,
                              hipStream_t stream) {
    (void)in_sizes; (void)n_in; (void)out_size; (void)ws_size;
    const float* inp = (const float*)d_in[0];   // [32,64,64,64] f32
    const float* emb = (const float*)d_in[1];   // [512,64] f32
    float* out    = (float*)d_out;              // out | loss | indices (concatenated)
    float* loss   = out + OUT_ELEMS;
    float* idxOut = out + OUT_ELEMS + 1;
    float* partial = (float*)d_ws;              // 1024 f32 partials

    vq_fused<<<NWG, 256, 0, stream>>>(inp, emb, out, idxOut, partial);
    vq_loss<<<1, 256, 0, stream>>>(partial, loss);
}
